// STRODE_80599356276802
// MI455X (gfx1250) — matrix operand order repaired and checked by reading, not compiled, not run on hardware
//
#include <hip/hip_runtime.h>
#include <cstdint>

// ---------------------------------------------------------------------------
// CDNA5 / gfx1250, wave32. GEMMs via V_WMMA_F32_16X16X32_F16.
// U=100 zero-padded to UP=128 (8 N-tiles, 4 K-tiles of 32).
// ---------------------------------------------------------------------------

typedef __attribute__((ext_vector_type(16))) _Float16 v16h;
typedef __attribute__((ext_vector_type(8)))  float    v8f;

#define WMMA16(a, b, c) \
  __builtin_amdgcn_wmma_f32_16x16x32_f16(false, (a), false, (b), (short)0, (c), false, false)

#define BATCH   8192
#define DDIM    512
#define UPAD    128     // padded hidden width
#define MB      64      // batch rows per block
#define WAVES   4       // MB/16
#define FRAG_DW 256     // dwords per 16x16x32 B-fragment (32 lanes * 8 dwords)

// ws fragment layout:
//   W1: frags [0,128)   f = kt*8 + nt   (kt<16 over K=512, nt<8 over N=128)
//   W2: frags [128,160) f = kt*8 + nt   (kt<4  over K=128, nt<8 over N=128)
//   W3: frags [160,288) f = nt*4 + kt   (nt<32 over N=512, kt<4 over K=128)
#define W1_FRAG_BASE 0
#define W2_FRAG_BASE (128 * FRAG_DW)
#define W3_FRAG_BASE (160 * FRAG_DW)

// ---------------------------------------------------------------------------
// tanh: prefer the CDNA5 hardware transcendental V_TANH_F32; otherwise a
// branch-free exp2/rcp approximation (no EXEC-mask churn between WMMA phases).
// ---------------------------------------------------------------------------
__device__ __forceinline__ float fast_tanhf(float x) {
#if __has_builtin(__builtin_amdgcn_tanhf)
  return __builtin_amdgcn_tanhf(x);
#else
  const float ax = __builtin_fabsf(x);
  // e = exp(-2|x|) computed as exp2(-2*log2(e)*|x|); underflows to 0 for big x.
  const float e  = __builtin_amdgcn_exp2f(-2.8853900817779268f * ax);
  const float t  = (1.0f - e) * __builtin_amdgcn_rcpf(1.0f + e);
  return __builtin_copysignf(t, x);
#endif
}

// ---------------------------------------------------------------------------
// Weight repack: f32 [N,K] row-major -> f16 WMMA B-fragments (B[k][n] = W[n][k])
// B-frag layout (16x16x32 f16): lanes 0-15: N=lane, K={0..7,16..23} (2 per VGPR);
//                               lanes 16-31: N=lane-16, K={8..15,24..31}.
// ---------------------------------------------------------------------------
__global__ void strode_prep_weights(const float* __restrict__ W1,
                                    const float* __restrict__ W2,
                                    const float* __restrict__ W3,
                                    uint32_t* __restrict__ frag) {
  const int f    = blockIdx.x;      // 0..287
  const int t    = threadIdx.x;     // 0..255
  const int lane = t >> 3;          // 0..31
  const int v    = t & 7;           // dword within lane
  const int nl   = lane & 15;
  const int kl   = (v < 4) ? ((lane >> 4) * 8) + 2 * v : 16 + ((lane >> 4) * 8) + 2 * (v - 4);   // K within 32-tile (even)

  float lo = 0.0f, hi = 0.0f;
  if (f < 128) {                                   // W1: [100,512], pad N
    const int kt = f >> 3, nt = f & 7;
    const int n = nt * 16 + nl, k0 = kt * 32 + kl;
    if (n < 100) { lo = W1[n * 512 + k0]; hi = W1[n * 512 + k0 + 1]; }
  } else if (f < 160) {                            // W2: [100,100], pad N and K
    const int g = f - 128;
    const int kt = g >> 3, nt = g & 7;
    const int n = nt * 16 + nl, k0 = kt * 32 + kl;
    if (n < 100) {
      if (k0     < 100) lo = W2[n * 100 + k0];
      if (k0 + 1 < 100) hi = W2[n * 100 + k0 + 1];
    }
  } else {                                         // W3: [512,100], pad K
    const int g = f - 160;
    const int nt = g >> 2, kt = g & 3;
    const int n = nt * 16 + nl, k0 = kt * 32 + kl;
    if (k0     < 100) lo = W3[n * 100 + k0];
    if (k0 + 1 < 100) hi = W3[n * 100 + k0 + 1];
  }
  union { _Float16 h[2]; uint32_t u; } pk;
  pk.h[0] = (_Float16)lo; pk.h[1] = (_Float16)hi;
  frag[f * FRAG_DW + lane * 8 + v] = pk.u;
}

// ---------------------------------------------------------------------------
// A-fragment builders. 16-bit A 16x32: lanes 0-15 (M=lane) hold K={0..7,16..23},
// lanes 16-31 (M=lane-16) hold K={8..15,24..31}; 2 halves per VGPR in K order.
// ---------------------------------------------------------------------------
__device__ __forceinline__ v16h a_from_f32(const float* __restrict__ rowp, int k0) {
  union { v16h v; _Float16 h[16]; } u;
#pragma unroll
  for (int j = 0; j < 8; ++j) {
    u.h[j]     = (_Float16)rowp[k0 + j];
    u.h[j + 8] = (_Float16)rowp[k0 + 16 + j];
  }
  return u.v;
}

__device__ __forceinline__ v16h a_from_f16(const _Float16* __restrict__ rowp, int k0) {
  union { v16h v; _Float16 h[16]; } u;
#pragma unroll
  for (int j = 0; j < 8; ++j) {
    u.h[j]     = rowp[k0 + j];
    u.h[j + 8] = rowp[k0 + 16 + j];
  }
  return u.v;
}

__device__ __forceinline__ v16h b_frag(const uint32_t* __restrict__ p) {
  union { uint4 q[2]; v16h v; } u;
  const uint4* q = (const uint4*)p;
  u.q[0] = q[0];
  u.q[1] = q[1];
  return u.v;
}

// ---------------------------------------------------------------------------
// Fused k-step Euler integrator. One wave integrates 16 batch rows.
// y kept in F32 in LDS for all k steps; h staged per-wave in LDS (F16) for the
// C-layout -> A-layout transpose between chained GEMMs.
// Occupancy is LDS-bound (144KB/block -> 2 blocks/WGP = 2 waves/SIMD), so
// allow up to ~512 VGPRs/wave (launch_bounds min-waves-per-EU = 2) to avoid
// scratch spills in the WMMA loops.
// ---------------------------------------------------------------------------
__global__ __launch_bounds__(WAVES * 32, 2) void strode_ode_main(
    const float* __restrict__ y0, const float* __restrict__ t_seq,
    const float* __restrict__ b1, const float* __restrict__ b2,
    const float* __restrict__ b3, const int* __restrict__ kptr,
    const uint32_t* __restrict__ frag, float* __restrict__ out) {
  __shared__ float    yT[MB * DDIM];          // 128 KB
  __shared__ _Float16 hT[WAVES][16 * UPAD];   // 16 KB

  const int tid      = threadIdx.x;
  const int wave     = tid >> 5;
  const int lane     = tid & 31;
  const int nl       = lane & 15;      // N column within tile / M row for A
  const int kh       = lane >> 4;      // lane-group (K-half / M-half select)
  const int waveRow0 = blockIdx.x * MB + wave * 16;
  const int kk       = *kptr;

  // Wave-private copy of its 16 rows of y0 into LDS (no cross-wave deps ever).
  {
    const float4* src = (const float4*)(y0 + (size_t)waveRow0 * DDIM);
    float4*       dst = (float4*)(yT + wave * 16 * DDIM);
    for (int i = lane; i < 16 * DDIM / 4; i += 32) dst[i] = src[i];
  }

  // dt per accumulator row r (C-layout: row m = r + 8*kh).
  float dtv[8];
#pragma unroll
  for (int r = 0; r < 8; ++r) {
    const int row = waveRow0 + r + 8 * kh;
    dtv[r] = (t_seq[2 * row + 1] - t_seq[2 * row]) / (float)kk;
  }
  // Small biases resident in VGPRs (padded region = 0). b3 is loaded per-tile.
  float b1v[8], b2v[8];
#pragma unroll
  for (int nt = 0; nt < 8; ++nt) {
    const int n = nt * 16 + nl;
    b1v[nt] = (n < 100) ? b1[n] : 0.0f;
    b2v[nt] = (n < 100) ? b2[n] : 0.0f;
  }

  const float*    yrow = yT + (wave * 16 + nl) * DDIM;  // A row for GEMM1
  _Float16*       hbuf = hT[wave];
  const _Float16* hrow = hbuf + nl * UPAD;              // A row for GEMM2/3
  const uint32_t* fW1  = frag + W1_FRAG_BASE;
  const uint32_t* fW2  = frag + W2_FRAG_BASE;
  const uint32_t* fW3  = frag + W3_FRAG_BASE;
  const int       lo8  = lane * 8;

  for (int s = 0; s < kk; ++s) {
    // ---- GEMM1: h1 = tanh(y[16x512] @ W1^T + b1) -> 16x128 -----------------
    v8f acc[8];
#pragma unroll
    for (int nt = 0; nt < 8; ++nt) acc[nt] = (v8f){0, 0, 0, 0, 0, 0, 0, 0};
#pragma clang loop unroll(disable)
    for (int kt = 0; kt < 16; ++kt) {
      const v16h a = a_from_f32(yrow, kt * 32 + 8 * kh);
#pragma unroll
      for (int nt = 0; nt < 8; ++nt) {
        const v16h b = b_frag(fW1 + (kt * 8 + nt) * FRAG_DW + lo8);
        acc[nt] = WMMA16(a, b, acc[nt]);
      }
    }
#pragma unroll
    for (int nt = 0; nt < 8; ++nt) {
#pragma unroll
      for (int r = 0; r < 8; ++r) {
        const float h = fast_tanhf(acc[nt][r] + b1v[nt]);
        hbuf[(r + 8 * kh) * UPAD + nt * 16 + nl] = (_Float16)h;  // C->LDS
      }
    }

    // ---- GEMM2: h2 = tanh(h1[16x128] @ W2^T + b2) -> 16x128 ----------------
    v8f acc2[8];
#pragma unroll
    for (int nt = 0; nt < 8; ++nt) acc2[nt] = (v8f){0, 0, 0, 0, 0, 0, 0, 0};
#pragma clang loop unroll(disable)
    for (int kt = 0; kt < 4; ++kt) {
      const v16h a = a_from_f16(hrow, kt * 32 + 8 * kh);
#pragma unroll
      for (int nt = 0; nt < 8; ++nt) {
        const v16h b = b_frag(fW2 + (kt * 8 + nt) * FRAG_DW + lo8);
        acc2[nt] = WMMA16(a, b, acc2[nt]);
      }
    }
#pragma unroll
    for (int nt = 0; nt < 8; ++nt) {
#pragma unroll
      for (int r = 0; r < 8; ++r) {
        const float h = fast_tanhf(acc2[nt][r] + b2v[nt]);
        hbuf[(r + 8 * kh) * UPAD + nt * 16 + nl] = (_Float16)h;  // overwrite h1
      }
    }

    // ---- GEMM3 + Euler update: y += (h2 @ W3^T + b3) * dt ------------------
    v16h ah[4];
#pragma unroll
    for (int kt = 0; kt < 4; ++kt) ah[kt] = a_from_f16(hrow, kt * 32 + 8 * kh);
#pragma clang loop unroll(disable)
    for (int nt = 0; nt < 32; ++nt) {
      v8f a3 = (v8f){0, 0, 0, 0, 0, 0, 0, 0};
#pragma unroll
      for (int kt = 0; kt < 4; ++kt) {
        const v16h b = b_frag(fW3 + (nt * 4 + kt) * FRAG_DW + lo8);
        a3 = WMMA16(ah[kt], b, a3);
      }
      const float b3g = b3[nt * 16 + nl];   // L2-resident, per-tile bias
#pragma unroll
      for (int r = 0; r < 8; ++r) {
        float* yp = yT + (wave * 16 + r + 8 * kh) * DDIM + nt * 16 + nl;
        *yp += (a3[r] + b3g) * dtv[r];
      }
    }
  }

  // ---- Write y_final into out[:, 0:512] (pitch 513) ------------------------
  for (int i = lane; i < 16 * DDIM; i += 32) {
    const int r = i >> 9, c = i & 511;
    out[(size_t)(waveRow0 + r) * (DDIM + 1) + c] = yT[(wave * 16 + r) * DDIM + c];
  }
}

// ---------------------------------------------------------------------------
// Scalar loss branch:
//   loss = ycum[-2] = EPS * sum_{t=1..98} g(m_t),  m_t = -1 + 0.01 t
//   g(m) = -bd/(m*log(-m)) * (Kv - log(-log(-m)))
//   Kv   = s + log(-bd + 0.01) - log(sd + 0.01)
// ---------------------------------------------------------------------------
__global__ void strode_loss(const float* __restrict__ bder,
                            const float* __restrict__ sphi,
                            const float* __restrict__ sder,
                            float* __restrict__ out) {
  const int b = blockIdx.x * blockDim.x + threadIdx.x;
  if (b >= BATCH) return;
  const float bd = bder[b];
  const float Kv = sphi[b] + logf(-bd + 0.01f) - logf(sder[b] + 0.01f);
  float sum = 0.0f;
  for (int t = 1; t <= 98; ++t) {
    const float m  = -1.0f + 0.01f * (float)t;
    const float lm = logf(-m);                       // < 0
    sum += (-bd) / (m * lm) * (Kv - logf(-lm));
  }
  out[(size_t)b * (DDIM + 1) + DDIM] = 0.01f * sum;
}

// ---------------------------------------------------------------------------
extern "C" void kernel_launch(void* const* d_in, const int* in_sizes, int n_in,
                              void* d_out, int out_size, void* d_ws, size_t ws_size,
                              hipStream_t stream) {
  const float* y0    = (const float*)d_in[0];
  const float* t_seq = (const float*)d_in[1];
  const float* W1    = (const float*)d_in[2];
  const float* b1    = (const float*)d_in[3];
  const float* W2    = (const float*)d_in[4];
  const float* b2    = (const float*)d_in[5];
  const float* W3    = (const float*)d_in[6];
  const float* b3    = (const float*)d_in[7];
  // d_in[8] (b_phi_zt) is unused by the reference loss.
  const float* bder  = (const float*)d_in[9];
  const float* sphi  = (const float*)d_in[10];
  const float* sder  = (const float*)d_in[11];
  const int*   kptr  = (const int*)d_in[12];
  float*       out   = (float*)d_out;
  uint32_t*    frag  = (uint32_t*)d_ws;   // 288 KB of f16 weight fragments

  strode_prep_weights<<<288, 256, 0, stream>>>(W1, W2, W3, frag);
  strode_ode_main<<<BATCH / MB, WAVES * 32, 0, stream>>>(y0, t_seq, b1, b2, b3,
                                                         kptr, frag, out);
  strode_loss<<<(BATCH + 255) / 256, 256, 0, stream>>>(bder, sphi, sder, out);
}
